// Shallow_Clue_Refinement_30605936951855
// MI455X (gfx1250) — matrix operand order repaired and checked by reading, not compiled, not run on hardware
//
#include <hip/hip_runtime.h>
#include <hip/hip_bf16.h>
#include <math.h>

// ---------------- CDNA5 types ----------------
typedef __bf16 bf16_t;
typedef bf16_t v16bf __attribute__((ext_vector_type(16)));
typedef float  v8f   __attribute__((ext_vector_type(8)));

// ---------------- problem constants (from reference setup_inputs) ----------
#define C_IN   256          // shallow channels
#define C_DEEP 512          // deep channels (2*C)
#define D_CH   32           // d = 2C/16
#define H_IN   128
#define W_IN   128
#define HW_IN  (H_IN * W_IN)     // 16384
#define H_OUT  64
#define W_OUT  64
#define HW_OUT (H_OUT * W_OUT)   // 4096
#define OUT_C  256

__device__ __forceinline__ int koff_a(int e, int hi) {
    // A 16x32 layout (contiguous, matches B): lanes 0-15 hold K {0..15},
    // lanes 16-31 hold K {16..31}; vector element e -> K = 16*hi + e.
    return 16 * hi + e;
}

// =====================================================================
// P0: pack an f32 weight matrix [rows x cols] into the CDNA5 WMMA bf16
//     A-operand lane layout: dst[((mt*nks + ks)*32 + lane)*16 + e]
//     holds A[mt*16 + lane%16][ks*32 + koff_a(e, lane/16)].
//     One thread per (mt, ks, lane); 16 bf16 stores each. One-time cost.
// =====================================================================
__global__ void __launch_bounds__(256)
p0_pack_a_bf16(const float* __restrict__ src, bf16_t* __restrict__ dst,
               int rows, int cols)
{
    const int t     = blockIdx.x * blockDim.x + threadIdx.x;
    const int nks   = cols >> 5;
    const int total = (rows >> 4) * nks * 32;
    if (t >= total) return;
    const int lane = t & 31;
    const int ks   = (t >> 5) % nks;
    const int mt   = t / (nks * 32);
    const int hi   = lane >> 4;
    const int m    = lane & 15;
    const float* __restrict__ r = src + (size_t)(mt * 16 + m) * cols + ks * 32;
    bf16_t* __restrict__ d = dst + (size_t)t * 16;
#pragma unroll
    for (int e = 0; e < 16; ++e) d[e] = (bf16_t)r[koff_a(e, hi)];
}

// =====================================================================
// K1: max_out[b,o,p] = sum_c KSA_w[o,c]*deep[b,c,p] + KSA_b[o]
//     M=32, K=512, N=4096. Each wave: 32x32 output tile
//     (2 m-subtiles x 2 n-subtiles = 4 WMMAs per k-step, 16 k-steps).
//     Block = 8 waves -> 256 columns; 16 blocks per batch.
// =====================================================================
__global__ void __launch_bounds__(256)
k1_deep_gemm(const float* __restrict__ deep,
             const bf16_t* __restrict__ apack,   // packed KSA_w (nks=16)
             const float* __restrict__ ksa_b,
             float* __restrict__ maxout)
{
    const int tid  = threadIdx.x;
    const int wave = tid >> 5;
    const int lane = tid & 31;
    const int b    = blockIdx.x >> 4;   // 16 blocks per batch
    const int nblk = blockIdx.x & 15;
    const int hi   = lane >> 4;
    const int n0   = nblk * 256 + wave * 32 + (lane & 15);  // n-subtile 0 col
    const int nks  = C_DEEP / 32;       // 16

    const float* __restrict__ dB = deep + (size_t)b * C_DEEP * HW_OUT;

    v8f acc00 = {}, acc01 = {};  // out-channels 0..15  x (n0, n0+16)
    v8f acc10 = {}, acc11 = {};  // out-channels 16..31 x (n0, n0+16)

    for (int ks = 0; ks < nks; ++ks) {
        const int k0 = ks * 32;
        const v16bf a0 = *(const v16bf*)(apack + ((size_t)(0 * nks + ks) * 32 + lane) * 16);
        const v16bf a1 = *(const v16bf*)(apack + ((size_t)(1 * nks + ks) * 32 + lane) * 16);
        v16bf bv0, bv1;
#pragma unroll
        for (int e = 0; e < 16; ++e) {
            // B 32x16 layout: element e holds K = 16*hi + e, col = lane%16
            const size_t rb = (size_t)(k0 + 16 * hi + e) * HW_OUT;
            bv0[e] = (bf16_t)dB[rb + n0];
            bv1[e] = (bf16_t)dB[rb + n0 + 16];
        }
        if (k0 + 32 < C_DEEP)   // gfx1250 global_prefetch_b8 on the B stream
            __builtin_prefetch(&dB[(size_t)(k0 + 32 + 16 * hi) * HW_OUT + n0], 0, 1);
        acc00 = __builtin_amdgcn_wmma_f32_16x16x32_bf16(false, a0, false, bv0, (short)0, acc00, false, false);
        acc01 = __builtin_amdgcn_wmma_f32_16x16x32_bf16(false, a0, false, bv1, (short)0, acc01, false, false);
        acc10 = __builtin_amdgcn_wmma_f32_16x16x32_bf16(false, a1, false, bv0, (short)0, acc10, false, false);
        acc11 = __builtin_amdgcn_wmma_f32_16x16x32_bf16(false, a1, false, bv1, (short)0, acc11, false, false);
    }

    float* __restrict__ mo = maxout + (size_t)b * D_CH * HW_OUT;
#pragma unroll
    for (int r = 0; r < 8; ++r) {
        const int o = r + 8 * hi;           // C/D layout: VGPR r -> row r (+8 for hi lanes)
        const float b0 = ksa_b[o], b1 = ksa_b[o + 16];
        mo[(size_t)o        * HW_OUT + n0]      = acc00[r] + b0;
        mo[(size_t)o        * HW_OUT + n0 + 16] = acc01[r] + b0;
        mo[(size_t)(o + 16) * HW_OUT + n0]      = acc10[r] + b1;
        mo[(size_t)(o + 16) * HW_OUT + n0 + 16] = acc11[r] + b1;
    }
}

// =====================================================================
// K2: per (b,p): q = max_d(max_out), qmean = mean_d(max_out)
// =====================================================================
__global__ void __launch_bounds__(256)
k2_reduce_q(const float* __restrict__ maxout,
            float* __restrict__ q,
            float* __restrict__ qmean,
            int B)
{
    const int idx = blockIdx.x * blockDim.x + threadIdx.x;
    if (idx >= B * HW_OUT) return;
    const int b = idx / HW_OUT;
    const int p = idx % HW_OUT;
    const float* __restrict__ mo = maxout + (size_t)b * D_CH * HW_OUT + p;
    float mx = -3.4e38f, sm = 0.f;
#pragma unroll
    for (int dd = 0; dd < D_CH; ++dd) {
        const float v = mo[(size_t)dd * HW_OUT];
        mx = fmaxf(mx, v);
        sm += v;
    }
    q[idx]     = mx;
    qmean[idx] = sm * (1.0f / D_CH);
}

// =====================================================================
// K3: lit[b,dd] = (1/hw) * sum_p q[b,p] * flat(max_out)[b][p*32 + dd]
//     (channel-scrambled torch .view(B,-1,d)).  Block per (b,dd).
// =====================================================================
__global__ void __launch_bounds__(256)
k3_lit(const float* __restrict__ maxout,
       const float* __restrict__ q,
       float* __restrict__ lit)
{
    __shared__ float red[256];
    const int b  = blockIdx.x >> 5;   // /32
    const int dd = blockIdx.x & 31;
    const float* __restrict__ flat = maxout + (size_t)b * D_CH * HW_OUT;
    const float* __restrict__ qb   = q + (size_t)b * HW_OUT;

    float s = 0.f;
    for (int p = threadIdx.x; p < HW_OUT; p += 256)
        s += qb[p] * flat[(size_t)p * D_CH + dd];
    red[threadIdx.x] = s;
    __syncthreads();
#pragma unroll
    for (int off = 128; off > 0; off >>= 1) {
        if (threadIdx.x < off) red[threadIdx.x] += red[threadIdx.x + off];
        __syncthreads();
    }
    if (threadIdx.x == 0)
        lit[b * D_CH + dd] = red[0] * (1.0f / HW_OUT);
}

// =====================================================================
// K4: re_score[b,p] = sum_dd lit[b,dd]*max_out[b,dd,p]; sig = sigmoid
// =====================================================================
__global__ void __launch_bounds__(256)
k4_rescore(const float* __restrict__ maxout,
           const float* __restrict__ lit,
           float* __restrict__ sig,
           float* __restrict__ rescore_out,
           int B)
{
    const int idx = blockIdx.x * blockDim.x + threadIdx.x;
    if (idx >= B * HW_OUT) return;
    const int b = idx / HW_OUT;
    const int p = idx % HW_OUT;
    const float* __restrict__ mo = maxout + (size_t)b * D_CH * HW_OUT + p;
    const float* __restrict__ lb = lit + b * D_CH;
    float s = 0.f;
#pragma unroll
    for (int dd = 0; dd < D_CH; ++dd)
        s += lb[dd] * mo[(size_t)dd * HW_OUT];
    rescore_out[idx] = s;
    sig[idx] = 1.0f / (1.0f + __expf(-s));
}

// =====================================================================
// K5: Kx[b,o,y,x] = sum_c K_w[o,c]*shallow[b,c,y,x] + K_b[o]
//     M=256, K=256, N=16384 per batch. Each wave: 32x32 output tile
//     (4 WMMAs per k-step, 8 k-steps). 8 waves per block.
// =====================================================================
__global__ void __launch_bounds__(256)
k5_shallow_gemm(const float* __restrict__ shallow,
                const bf16_t* __restrict__ apack,   // packed K_w (nks=8)
                const float* __restrict__ k_b,
                float* __restrict__ kx)
{
    const int lane = threadIdx.x & 31;
    const int gw   = blockIdx.x * 8 + (threadIdx.x >> 5);
    const int b    = gw >> 12;              // 4096 waves per batch
    const int rem  = gw & 4095;
    const int mt2  = rem >> 9;              // 32-row tile 0..7
    const int nt2  = rem & 511;             // 32-col tile 0..511
    const int hi   = lane >> 4;
    const int n0   = nt2 * 32 + (lane & 15);
    const int nks  = C_IN / 32;             // 8

    const float* __restrict__ sB = shallow + (size_t)b * C_IN * HW_IN;

    v8f acc00 = {}, acc01 = {};  // rows mt2*32+{0..15}  x (n0, n0+16)
    v8f acc10 = {}, acc11 = {};  // rows mt2*32+{16..31} x (n0, n0+16)

    for (int ks = 0; ks < nks; ++ks) {
        const int k0 = ks * 32;
        const v16bf a0 = *(const v16bf*)(apack + ((size_t)((mt2 * 2 + 0) * nks + ks) * 32 + lane) * 16);
        const v16bf a1 = *(const v16bf*)(apack + ((size_t)((mt2 * 2 + 1) * nks + ks) * 32 + lane) * 16);
        v16bf bv0, bv1;
#pragma unroll
        for (int e = 0; e < 16; ++e) {
            const size_t rb = (size_t)(k0 + 16 * hi + e) * HW_IN;
            bv0[e] = (bf16_t)sB[rb + n0];
            bv1[e] = (bf16_t)sB[rb + n0 + 16];
        }
        if (k0 + 32 < C_IN)   // gfx1250 global_prefetch_b8 on the B stream
            __builtin_prefetch(&sB[(size_t)(k0 + 32 + 16 * hi) * HW_IN + n0], 0, 1);
        acc00 = __builtin_amdgcn_wmma_f32_16x16x32_bf16(false, a0, false, bv0, (short)0, acc00, false, false);
        acc01 = __builtin_amdgcn_wmma_f32_16x16x32_bf16(false, a0, false, bv1, (short)0, acc01, false, false);
        acc10 = __builtin_amdgcn_wmma_f32_16x16x32_bf16(false, a1, false, bv0, (short)0, acc10, false, false);
        acc11 = __builtin_amdgcn_wmma_f32_16x16x32_bf16(false, a1, false, bv1, (short)0, acc11, false, false);
    }

    float* __restrict__ kxb = kx + (size_t)b * OUT_C * HW_IN;
#pragma unroll
    for (int r = 0; r < 8; ++r) {
        const int o0 = mt2 * 32 + r + 8 * hi;
        const int o1 = o0 + 16;
        const float b0 = k_b[o0], b1 = k_b[o1];
        kxb[(size_t)o0 * HW_IN + n0]      = acc00[r] + b0;
        kxb[(size_t)o0 * HW_IN + n0 + 16] = acc01[r] + b0;
        kxb[(size_t)o1 * HW_IN + n0]      = acc10[r] + b1;
        kxb[(size_t)o1 * HW_IN + n0 + 16] = acc11[r] + b1;
    }
}

// =====================================================================
// K6: fused unfold(3x3, stride 2, pad 1) + softmax attention + scale.
//     out[b,o,p] = att * (1 + sig[b,p]);  att = sum_kk softmax(qmean*v)_kk * v_kk
// =====================================================================
__global__ void __launch_bounds__(256)
k6_attention(const float* __restrict__ kx,
             const float* __restrict__ qmean,
             const float* __restrict__ sig,
             float* __restrict__ out,
             int B)
{
    const int idx = blockIdx.x * blockDim.x + threadIdx.x;
    if (idx >= B * OUT_C * HW_OUT) return;
    const int b  = idx / (OUT_C * HW_OUT);
    const int o  = (idx / HW_OUT) % OUT_C;
    const int p  = idx % HW_OUT;
    const int ph = p >> 6;   // /64
    const int pw = p & 63;

    const float qm = qmean[b * HW_OUT + p];
    const float* __restrict__ kxo = kx + ((size_t)b * OUT_C + o) * HW_IN;

    float v[9];
#pragma unroll
    for (int i = 0; i < 3; ++i) {
        const int y = 2 * ph - 1 + i;
#pragma unroll
        for (int j = 0; j < 3; ++j) {
            const int x = 2 * pw - 1 + j;
            const bool ok = (y >= 0) & (y < H_IN) & (x >= 0) & (x < W_IN);
            v[i * 3 + j] = ok ? kxo[(size_t)y * W_IN + x] : 0.0f;
        }
    }

    float mx = -3.4e38f;
#pragma unroll
    for (int kk = 0; kk < 9; ++kk) mx = fmaxf(mx, qm * v[kk]);
    float den = 0.f, num = 0.f;
#pragma unroll
    for (int kk = 0; kk < 9; ++kk) {
        const float e = __expf(qm * v[kk] - mx);
        den += e;
        num += e * v[kk];
    }
    const float att = num / den;
    out[idx] = att * (1.0f + sig[b * HW_OUT + p]);
}

// =====================================================================
// host launch
// =====================================================================
extern "C" void kernel_launch(void* const* d_in, const int* in_sizes, int n_in,
                              void* d_out, int out_size, void* d_ws, size_t ws_size,
                              hipStream_t stream)
{
    const float* shallow = (const float*)d_in[0];
    const float* deep    = (const float*)d_in[1];
    const float* K_w     = (const float*)d_in[2];
    const float* K_b     = (const float*)d_in[3];
    const float* KSA_w   = (const float*)d_in[4];
    const float* KSA_b   = (const float*)d_in[5];

    const int B = in_sizes[0] / (C_IN * HW_IN);   // 4 for reference shapes

    // workspace layout (floats)
    float* ws = (float*)d_ws;
    size_t off = 0;
    float* maxout = ws + off; off += (size_t)B * D_CH * HW_OUT;   // [B,32,4096]
    float* q      = ws + off; off += (size_t)B * HW_OUT;          // [B,4096]
    float* qmean  = ws + off; off += (size_t)B * HW_OUT;          // [B,4096]
    float* lit    = ws + off; off += (size_t)B * D_CH;            // [B,32]
    float* sig    = ws + off; off += (size_t)B * HW_OUT;          // [B,4096]
    float* kx     = ws + off; off += (size_t)B * OUT_C * HW_IN;   // [B,256,16384]

    // bf16 packed weights (64B-aligned)
    off = (off + 15) & ~(size_t)15;
    bf16_t* apackK   = (bf16_t*)(ws + off);   // 16 mt * 8 ks * 32 * 16 = 65536 bf16
    off += 65536 / 2;
    off = (off + 15) & ~(size_t)15;
    bf16_t* apackKSA = (bf16_t*)(ws + off);   // 2 mt * 16 ks * 32 * 16 = 16384 bf16
    off += 16384 / 2;

    float* out_main = (float*)d_out;                          // [B,256,64,64]
    float* out_rs   = out_main + (size_t)B * OUT_C * HW_OUT;  // [B,1,64,64]

    // P0: pack weights into WMMA A-operand bf16 layout (one-time)
    p0_pack_a_bf16<<<(16 * 8 * 32 + 255) / 256, 256, 0, stream>>>(K_w, apackK, OUT_C, C_IN);
    p0_pack_a_bf16<<<(2 * 16 * 32 + 255) / 256, 256, 0, stream>>>(KSA_w, apackKSA, D_CH, C_DEEP);

    // K1: deep GEMM -> max_out  (32x32 tile per wave)
    k1_deep_gemm<<<B * 16, 256, 0, stream>>>(deep, apackKSA, KSA_b, maxout);

    // K2: q / qmean
    k2_reduce_q<<<(B * HW_OUT + 255) / 256, 256, 0, stream>>>(maxout, q, qmean, B);

    // K3: lit (scrambled-view GEMV)
    k3_lit<<<B * D_CH, 256, 0, stream>>>(maxout, q, lit);

    // K4: re_score + sigmoid
    k4_rescore<<<(B * HW_OUT + 255) / 256, 256, 0, stream>>>(maxout, lit, sig, out_rs, B);

    // K5: shallow GEMM -> Kx  (32x32 tile per wave)
    k5_shallow_gemm<<<B * 512, 256, 0, stream>>>(shallow, apackK, K_b, kx);

    // K6: fused unfold + softmax attention + (1+sig) scale
    k6_attention<<<(B * OUT_C * HW_OUT + 255) / 256, 256, 0, stream>>>(kx, qmean, sig, out_main, B);

    (void)n_in; (void)out_size; (void)ws_size;
}
